// MCAB_1357209666377
// MI455X (gfx1250) — matrix operand order repaired and checked by reading, not compiled, not run on hardware
//
#include <hip/hip_runtime.h>
#include <hip/hip_bf16.h>

// ---------------------------------------------------------------------------
// MCAB stereo cross-attention, CDNA5 (gfx1250) WMMA implementation.
// Shapes: B=4, C=128, H=96, W=256.  All GEMMs via v_wmma_f32_16x16x32_f16.
// V-tile staging in the attention kernel uses the Tensor Data Mover (TDM) or
// async global->LDS copies when the toolchain exposes them, overlapped with
// the softmax tail.
// ---------------------------------------------------------------------------

#define Cc   128
#define C2   256
#define Hh   96
#define Ww   256
#define HW   24576           // Hh*Ww
#define PB   24576           // pixels per batch-plane
#define NPIX 98304           // B*HW
#define NOUT 12582912        // B*C*H*W

typedef __attribute__((ext_vector_type(16))) _Float16     v16h;
typedef __attribute__((ext_vector_type(8)))  float        v8f;
typedef __attribute__((ext_vector_type(4)))  unsigned int v4u;
typedef __attribute__((ext_vector_type(4)))  int          v4i;
typedef __attribute__((ext_vector_type(8)))  int          v8i;

#define DEVINL __device__ __forceinline__

// ---- CDNA5 data-movement feature detection --------------------------------
#if __has_builtin(__builtin_amdgcn_tensor_load_to_lds) && \
    __has_builtin(__builtin_amdgcn_s_wait_tensorcnt)
#define V_COPY_TDM 1
#elif __has_builtin(__builtin_amdgcn_global_load_async_to_lds_b128)
#define V_COPY_ASYNC 1
#endif

DEVINL v8f wmma_f16(v16h a, v16h b, v8f c) {
  // 8 args: (neg_a, A, neg_b, B, c_mod, C, reuse_a, reuse_b)
  return __builtin_amdgcn_wmma_f32_16x16x32_f16(false, a, false, b, (short)0, c,
                                                false, false);
}

DEVINL void vz8(v8f& v) {
#pragma unroll
  for (int i = 0; i < 8; ++i) v[i] = 0.0f;
}

// A-fragment from a row-major [M][K] f16 LDS tile (16x32 A layout, ISA 7.12.2):
// lane 0-15  hold M=lane,    K = k0+{0..7, 16..23}
// lane 16-31 hold M=lane-16, K = k0+{8..15, 24..31}
DEVINL v16h frag_row(const _Float16* p, int r0, int k0, int ld, int lane) {
  const _Float16* q = p + (r0 + (lane & 15)) * ld + k0 + ((lane >> 4) << 3);
  v16h a;
#pragma unroll
  for (int i = 0; i < 8; ++i) { a[i] = q[i]; a[i + 8] = q[i + 16]; }
  return a;
}

// B-fragment from a K-contiguous [N][K] f16 LDS tile (32x16 B layout):
// lane 0-15  hold N=lane,    K = k0+{0..7, 16..23}
// lane 16-31 hold N=lane-16, K = k0+{8..15, 24..31}
DEVINL v16h frag_kc(const _Float16* p, int n0, int k0, int ld, int lane) {
  const _Float16* q = p + (n0 + (lane & 15)) * ld + k0 + ((lane >> 4) << 3);
  v16h a;
#pragma unroll
  for (int i = 0; i < 8; ++i) { a[i] = q[i]; a[i + 8] = q[i + 16]; }
  return a;
}

#if defined(V_COPY_TDM)
// One 2D TDM descriptor: rows x cols f16 tile, global row stride in elements,
// dense in LDS at byte offset lds_off.  CDNA5 ISA.
DEVINL void tdm_load_tile_f16(const void* gaddr, unsigned lds_off,
                              unsigned rows, unsigned cols,
                              unsigned row_stride_elems) {
  unsigned long long ga = (unsigned long long)(uintptr_t)gaddr;
  v4u g0;
  g0[0] = 1u;                                        // count=1, user descriptor
  g0[1] = lds_off;                                   // lds_addr (bytes)
  g0[2] = (unsigned)(ga & 0xFFFFFFFFu);              // global_addr[31:0]
  g0[3] = (unsigned)((ga >> 32) & 0x1FFFFFFu) | (2u << 30);  // [56:32] | type=2
  v8i g1;
  g1[0] = (int)(1u << 16);                           // data_size=1 (2 bytes)
  g1[1] = (int)((cols & 0xFFFFu) << 16);             // tensor_dim0[15:0]
  g1[2] = (int)(((cols >> 16) & 0xFFFFu) | ((rows & 0xFFFFu) << 16));
  g1[3] = (int)(((rows >> 16) & 0xFFFFu) | ((cols & 0xFFFFu) << 16)); // tile_dim0
  g1[4] = (int)(rows & 0xFFFFu);                     // tile_dim1 (tile_dim2=0)
  g1[5] = (int)row_stride_elems;                     // tensor_dim0_stride[31:0]
  g1[6] = 0;
  g1[7] = 0;
  v4i z4; z4[0] = 0; z4[1] = 0; z4[2] = 0; z4[3] = 0;
#if __clang_major__ >= 23
  v8i z8; for (int i = 0; i < 8; ++i) z8[i] = 0;
  __builtin_amdgcn_tensor_load_to_lds(g0, g1, z4, z4, z8, 0);
#else
  __builtin_amdgcn_tensor_load_to_lds(g0, g1, z4, z4, 0);
#endif
}
#endif

#if defined(V_COPY_ASYNC)
DEVINL void async_b128(const _Float16* g, _Float16* l) {
  __builtin_amdgcn_global_load_async_to_lds_b128(
      (__attribute__((address_space(1))) void*)g,
      (__attribute__((address_space(3))) void*)l, 0, 0);
}
#endif

// ---------------------------------------------------------------------------
// K1: pre[b,o,p] = sum_c W[o,c] * x[b,c,p]   (o < 256, c < 128), f16 output.
// One block = 64 pixels x all 256 output rows.  8 waves, 32 rows each.
// ---------------------------------------------------------------------------
__global__ void k_conv1x1(const float* __restrict__ x,
                          const float* __restrict__ Wm,
                          _Float16* __restrict__ pre) {
  extern __shared__ char smem[];
  _Float16* Wl = (_Float16*)smem;          // [256][128] row-major (o,c)
  _Float16* Xt = Wl + 256 * 128;           // [64][128]  (pixel-major, K-contig)
  const int tid = threadIdx.x, lane = tid & 31, wid = tid >> 5;
  const int b  = blockIdx.x / (PB / 64);
  const int p0 = (blockIdx.x % (PB / 64)) * 64;

  for (int i = tid; i < 256 * 128; i += 256) Wl[i] = (_Float16)Wm[i];
  for (int i = tid; i < 128 * 64; i += 256) {
    int c = i >> 6, n = i & 63;
    Xt[n * 128 + c] = (_Float16)x[(b * Cc + c) * HW + p0 + n];
  }
  __syncthreads();

  const int m0 = wid * 32;
  v8f acc[2][4];
#pragma unroll
  for (int i = 0; i < 2; ++i)
#pragma unroll
    for (int j = 0; j < 4; ++j) vz8(acc[i][j]);

#pragma unroll
  for (int kk = 0; kk < 4; ++kk) {
    const int k0 = kk * 32;
    v16h a0 = frag_row(Wl, m0,      k0, 128, lane);
    v16h a1 = frag_row(Wl, m0 + 16, k0, 128, lane);
#pragma unroll
    for (int nt = 0; nt < 4; ++nt) {
      v16h bf = frag_kc(Xt, nt * 16, k0, 128, lane);
      acc[0][nt] = wmma_f16(a0, bf, acc[0][nt]);
      acc[1][nt] = wmma_f16(a1, bf, acc[1][nt]);
    }
  }

  const int hi = lane >> 4, nl = lane & 15;
#pragma unroll
  for (int mt = 0; mt < 2; ++mt)
#pragma unroll
    for (int nt = 0; nt < 4; ++nt)
#pragma unroll
      for (int vr = 0; vr < 8; ++vr) {
        int row = m0 + mt * 16 + vr + 8 * hi;
        pre[(b * C2 + row) * HW + p0 + nt * 16 + nl] = (_Float16)acc[mt][nt][vr];
      }
}

// ---------------------------------------------------------------------------
// K2: depthwise 3x3, SAME zero padding.  pre(f16) -> fus(f16).
// ---------------------------------------------------------------------------
__global__ void k_dwconv(const _Float16* __restrict__ pre,
                         const float* __restrict__ kw,
                         _Float16* __restrict__ fus) {
  int idx = blockIdx.x * 256 + threadIdx.x;        // < B*C2*HW = 25165824
  int p  = idx % HW;
  int bc = idx / HW;                               // b*256 + ch
  int ch = bc & 255;
  int hh = p >> 8, ww = p & 255;
  const float* kp = kw + ch * 9;
  float s = 0.0f;
#pragma unroll
  for (int dy = -1; dy <= 1; ++dy) {
    int h2 = hh + dy;
    if (h2 < 0 || h2 >= Hh) continue;
#pragma unroll
    for (int dx = -1; dx <= 1; ++dx) {
      int w2 = ww + dx;
      if (w2 < 0 || w2 >= Ww) continue;
      s += kp[(dy + 1) * 3 + (dx + 1)] * (float)pre[bc * HW + h2 * Ww + w2];
    }
  }
  fus[idx] = (_Float16)s;
}

// ---------------------------------------------------------------------------
// K3: Q = Wq * LayerNorm_C(mid) + bq, where mid = fus[:, C:2C].  f16 output.
// One block = 64 pixels.  LN stats per pixel over the 128-channel row.
// ---------------------------------------------------------------------------
__global__ void k_ln_qproj(const _Float16* __restrict__ fus,
                           const float* __restrict__ lnw,
                           const float* __restrict__ lnb,
                           const float* __restrict__ Wq,
                           const float* __restrict__ bq,
                           _Float16* __restrict__ Q) {
  extern __shared__ char smem[];
  _Float16* Wl = (_Float16*)smem;          // [128][128]
  _Float16* Xt = Wl + 128 * 128;           // [64][128] pixel-major
  const int tid = threadIdx.x, lane = tid & 31, wid = tid >> 5;
  const int b  = blockIdx.x / (PB / 64);
  const int p0 = (blockIdx.x % (PB / 64)) * 64;

  for (int i = tid; i < 128 * 128; i += 256) Wl[i] = (_Float16)Wq[i];
  for (int i = tid; i < 128 * 64; i += 256) {
    int c = i >> 6, n = i & 63;
    Xt[n * 128 + c] = fus[(b * C2 + Cc + c) * HW + p0 + n];
  }
  __syncthreads();

  if (tid < 64) {                          // one thread per pixel: LN stats
    _Float16* row = Xt + tid * 128;
    float s = 0.0f, sq = 0.0f;
#pragma unroll 8
    for (int c = 0; c < 128; ++c) { float v = (float)row[c]; s += v; sq += v * v; }
    float mu = s * (1.0f / 128.0f);
    float var = sq * (1.0f / 128.0f) - mu * mu;
    float rstd = rsqrtf(var + 1e-5f);
#pragma unroll 8
    for (int c = 0; c < 128; ++c) {
      float v = ((float)row[c] - mu) * rstd * lnw[c] + lnb[c];
      row[c] = (_Float16)v;
    }
  }
  __syncthreads();

  const int m0 = wid * 16;
  v8f acc[4];
#pragma unroll
  for (int j = 0; j < 4; ++j) vz8(acc[j]);
#pragma unroll
  for (int kk = 0; kk < 4; ++kk) {
    const int k0 = kk * 32;
    v16h a = frag_row(Wl, m0, k0, 128, lane);
#pragma unroll
    for (int nt = 0; nt < 4; ++nt) {
      v16h bf = frag_kc(Xt, nt * 16, k0, 128, lane);
      acc[nt] = wmma_f16(a, bf, acc[nt]);
    }
  }

  const int hi = lane >> 4, nl = lane & 15;
#pragma unroll
  for (int vr = 0; vr < 8; ++vr) {
    int row = m0 + vr + 8 * hi;
    float bias = bq[row];
#pragma unroll
    for (int nt = 0; nt < 4; ++nt)
      Q[(b * Cc + row) * HW + p0 + nt * 16 + nl] = (_Float16)(acc[nt][vr] + bias);
  }
}

// ---------------------------------------------------------------------------
// K4: fused row attention for one (b, h, 64-wide query block):
//   S = scale * Qa^T Qb            (64 x 256, K = 128)
//   P = softmax_rows(S)            (kept in LDS as f16)
//   O = V * P^T                    (128 x 64, K = 256)
//   out = x + O * scv              (f32 residual write)
// The V tile is DMA'd into the retired Qb LDS slot via TDM / async-to-LDS,
// overlapped with the softmax tail.
// ---------------------------------------------------------------------------
__global__ void k_attn(const _Float16* __restrict__ Qa,
                       const _Float16* __restrict__ Qb,
                       const _Float16* __restrict__ Vb,
                       const float* __restrict__ x,
                       const float* __restrict__ scv,
                       float* __restrict__ out) {
  extern __shared__ char smem[];
  _Float16* Qa_s = (_Float16*)smem;            // [64][128]  query-major
  _Float16* Qb_s = Qa_s + 64 * 128;            // [256][128] key-major (reused as V)
  _Float16* V_s  = Qb_s;                       // [128][256] chan-major (alias)
  _Float16* P_s  = Qb_s + 256 * 128;           // [64][256]  query-major
  float*    red  = (float*)(P_s + 64 * 256);   // [2][64]

  const int tid = threadIdx.x, lane = tid & 31, wid = tid >> 5;
  const int w0 = blockIdx.x * 64;
  const int h  = blockIdx.y;
  const int b  = blockIdx.z;
  const int plane = h * Ww;
  const _Float16* vsrc = Vb + (b * C2) * HW + plane;   // V tile row 0

  // Hint the later V tile toward the caches while we chew on S.
  {
    int c = tid >> 1, off = (tid & 1) * 128;
    __builtin_prefetch(&vsrc[c * HW + off], 0, 3);
  }

  for (int i = tid; i < 64 * 128; i += 256) {      // Qa block, transposed store
    int k = i >> 6, m = i & 63;
    Qa_s[m * 128 + k] = Qa[(b * Cc + k) * HW + plane + w0 + m];
  }
  for (int i = tid; i < 256 * 128; i += 256) {     // full Qb row, transposed
    int k = i >> 8, n = i & 255;
    Qb_s[n * 128 + k] = Qb[(b * Cc + k) * HW + plane + n];
  }
  __syncthreads();

  // ---- S = Qa^T Qb : wave -> 16 query rows x 128 key cols -----------------
  const int band = wid >> 1;          // 0..3 : query-row band
  const int ch   = wid & 1;           // 0..1 : key-column half
  const int m0   = band * 16;
  const int nbase = ch * 128;
  const int hi = lane >> 4, nl = lane & 15;

  v8f s[8];
#pragma unroll
  for (int nt = 0; nt < 8; ++nt) vz8(s[nt]);
#pragma unroll
  for (int kk = 0; kk < 4; ++kk) {
    const int k0 = kk * 32;
    v16h a = frag_row(Qa_s, m0, k0, 128, lane);
#pragma unroll
    for (int nt = 0; nt < 8; ++nt) {
      v16h bf = frag_kc(Qb_s, nbase + nt * 16, k0, 128, lane);
      s[nt] = wmma_f16(a, bf, s[nt]);
    }
  }

  const float scale = 0.0883883476483184f;   // 128^-0.5
#pragma unroll
  for (int nt = 0; nt < 8; ++nt)
#pragma unroll
    for (int vr = 0; vr < 8; ++vr) s[nt][vr] *= scale;

  // ---- softmax over the 256 key columns, per query row --------------------
  const int rbase = m0 + 8 * hi;             // this lane's 8 rows: rbase+vr
  float mx[8];
#pragma unroll
  for (int vr = 0; vr < 8; ++vr) {
    float m = s[0][vr];
#pragma unroll
    for (int nt = 1; nt < 8; ++nt) m = fmaxf(m, s[nt][vr]);
    m = fmaxf(m, __shfl_xor(m, 8, 32));      // reduce inside 16-lane half
    m = fmaxf(m, __shfl_xor(m, 4, 32));
    m = fmaxf(m, __shfl_xor(m, 2, 32));
    m = fmaxf(m, __shfl_xor(m, 1, 32));
    mx[vr] = m;
  }
  if (nl == 0) {
#pragma unroll
    for (int vr = 0; vr < 8; ++vr) red[ch * 64 + rbase + vr] = mx[vr];
  }
  __syncthreads();      // maxima published; all waves done reading Qb_s

  // ---- V tile DMA into the retired Qb slot, overlapping the softmax tail --
#if defined(V_COPY_TDM)
  if (wid == 0 && lane == 0)
    tdm_load_tile_f16(vsrc, (unsigned)(64 * 128 * sizeof(_Float16)),
                      /*rows=*/128, /*cols=*/256, /*row_stride=*/HW);
#elif defined(V_COPY_ASYNC)
  for (int i = tid; i < 128 * 32; i += 256) {      // 16-byte chunks
    int c = i >> 5, q = (i & 31) * 8;
    async_b128(vsrc + c * HW + q, V_s + c * 256 + q);
  }
#endif

  float gmx[8];
#pragma unroll
  for (int vr = 0; vr < 8; ++vr)
    gmx[vr] = fmaxf(red[rbase + vr], red[64 + rbase + vr]);

  float sm[8];
#pragma unroll
  for (int vr = 0; vr < 8; ++vr) {
    float t = 0.0f;
#pragma unroll
    for (int nt = 0; nt < 8; ++nt) {
      float e = __expf(s[nt][vr] - gmx[vr]);
      s[nt][vr] = e;
      t += e;
    }
    t += __shfl_xor(t, 8, 32);
    t += __shfl_xor(t, 4, 32);
    t += __shfl_xor(t, 2, 32);
    t += __shfl_xor(t, 1, 32);
    sm[vr] = t;
  }
  __syncthreads();                            // everyone done reading maxima
  if (nl == 0) {
#pragma unroll
    for (int vr = 0; vr < 8; ++vr) red[ch * 64 + rbase + vr] = sm[vr];
  }
  __syncthreads();

#pragma unroll
  for (int vr = 0; vr < 8; ++vr) {
    int row = rbase + vr;
    float rinv = 1.0f / (red[row] + red[64 + row]);
#pragma unroll
    for (int nt = 0; nt < 8; ++nt)
      P_s[row * 256 + nbase + nt * 16 + nl] = (_Float16)(s[nt][vr] * rinv);
  }
  __syncthreads();                            // P published

  // ---- make the V tile resident in LDS ------------------------------------
#if defined(V_COPY_TDM)
  if (wid == 0) __builtin_amdgcn_s_wait_tensorcnt(0);
#elif defined(V_COPY_ASYNC)
#if __has_builtin(__builtin_amdgcn_s_wait_asynccnt)
  __builtin_amdgcn_s_wait_asynccnt(0);
#else
  asm volatile("s_wait_asynccnt 0" ::: "memory");
#endif
#else
  for (int i = tid; i < 128 * 256; i += 256) {
    int c = i >> 8, v = i & 255;
    V_s[c * 256 + v] = vsrc[c * HW + v];
  }
#endif
  __syncthreads();

  // ---- O = V * P^T : wave -> 16 channel rows x 64 query cols --------------
  const int c0 = wid * 16;
  v8f o[4];
#pragma unroll
  for (int nt = 0; nt < 4; ++nt) vz8(o[nt]);
#pragma unroll
  for (int kk = 0; kk < 8; ++kk) {
    const int k0 = kk * 32;
    v16h a = frag_row(V_s, c0, k0, 256, lane);
#pragma unroll
    for (int nt = 0; nt < 4; ++nt) {
      v16h bf = frag_kc(P_s, nt * 16, k0, 256, lane);
      o[nt] = wmma_f16(a, bf, o[nt]);
    }
  }

#pragma unroll
  for (int vr = 0; vr < 8; ++vr) {
    int c = c0 + vr + 8 * hi;
    float g = scv[c];
#pragma unroll
    for (int nt = 0; nt < 4; ++nt) {
      int idx = (b * Cc + c) * HW + plane + w0 + nt * 16 + nl;
      out[idx] = x[idx] + o[nt][vr] * g;
    }
  }
}

// ---------------------------------------------------------------------------
extern "C" void kernel_launch(void* const* d_in, const int* in_sizes, int n_in,
                              void* d_out, int out_size, void* d_ws, size_t ws_size,
                              hipStream_t stream) {
  (void)in_sizes; (void)n_in; (void)out_size; (void)ws_size;
  const float* x_l  = (const float*)d_in[0];
  const float* x_r  = (const float*)d_in[1];
  const float* Wl_m = (const float*)d_in[2];
  const float* Wr_m = (const float*)d_in[3];
  const float* dw_l = (const float*)d_in[4];
  const float* dw_r = (const float*)d_in[5];
  const float* lnl_w = (const float*)d_in[6];
  const float* lnl_b = (const float*)d_in[7];
  const float* lnr_w = (const float*)d_in[8];
  const float* lnr_b = (const float*)d_in[9];
  const float* Wq_l = (const float*)d_in[10];
  const float* bq_l = (const float*)d_in[11];
  const float* Wq_r = (const float*)d_in[12];
  const float* bq_r = (const float*)d_in[13];
  const float* beta  = (const float*)d_in[14];
  const float* gamma = (const float*)d_in[15];

  // f16 workspace layout
  _Float16* ws = (_Float16*)d_ws;
  const size_t SZ2 = (size_t)4 * C2 * HW;   // 25165824 (B x 2C x H x W)
  const size_t SZ1 = (size_t)4 * Cc * HW;   // 12582912 (B x C x H x W)
  _Float16* pre_l = ws;
  _Float16* pre_r = pre_l + SZ2;
  _Float16* fus_l = pre_r + SZ2;
  _Float16* fus_r = fus_l + SZ2;
  _Float16* Q_l   = fus_r + SZ2;
  _Float16* Q_r   = Q_l + SZ1;

  float* out_l = (float*)d_out;
  float* out_r = out_l + NOUT;

  static bool attr_set = false;  // host-side, deterministic, not stream state
  if (!attr_set) {
    hipFuncSetAttribute((const void*)k_conv1x1,
                        hipFuncAttributeMaxDynamicSharedMemorySize, 81920);
    hipFuncSetAttribute((const void*)k_attn,
                        hipFuncAttributeMaxDynamicSharedMemorySize, 115200);
    attr_set = true;
  }

  const int nblk64 = NPIX / 64;             // 1536

  // 1) 1x1 conv GEMMs (C -> 2C)
  k_conv1x1<<<nblk64, 256, 81920, stream>>>(x_l, Wl_m, pre_l);
  k_conv1x1<<<nblk64, 256, 81920, stream>>>(x_r, Wr_m, pre_r);

  // 2) depthwise 3x3
  const int ndw = (int)(SZ2 / 256);         // 98304 blocks
  k_dwconv<<<ndw, 256, 0, stream>>>(pre_l, dw_l, fus_l);
  k_dwconv<<<ndw, 256, 0, stream>>>(pre_r, dw_r, fus_r);

  // 3) LayerNorm + Q projection
  k_ln_qproj<<<nblk64, 256, 49152, stream>>>(fus_l, lnl_w, lnl_b, Wq_l, bq_l, Q_l);
  k_ln_qproj<<<nblk64, 256, 49152, stream>>>(fus_r, lnr_w, lnr_b, Wq_r, bq_r, Q_r);

  // 4) fused attention. V_{l,r} are the first C channels of fus_{l,r}.
  dim3 ag(Ww / 64, Hh, 4);
  // F_r2l: rows=w, softmax over v, contract V_r  -> out_l
  k_attn<<<ag, 256, 115200, stream>>>(Q_l, Q_r, fus_r, x_l, beta, out_l);
  // F_l2r: attn^T = Q_r^T Q_l, softmax over w, contract V_l -> out_r
  k_attn<<<ag, 256, 115200, stream>>>(Q_r, Q_l, fus_l, x_r, gamma, out_r);
}
